// GNNModel_55576876810859
// MI455X (gfx1250) — matrix operand order repaired and checked by reading, not compiled, not run on hardware
//
#include <hip/hip_runtime.h>

// ---------------------------------------------------------------------------
// CDNA5 (gfx1250) hetero-GraphSAGE forward.
//  - scatter/mean phases: plain VMEM + global_atomic_add_f32 (bandwidth bound)
//  - all GEMMs: v_wmma_f32_16x16x32_f16 with split-f16 (hi/lo) operands for
//    ~fp32 accuracy; 3 WMMAs per 16x16x32 step. Compute is far from the
//    bottleneck (AI ~ 100 FLOP/B vs 23.3 TB/s HBM), so the extra WMMAs cost
//    nothing and preserve the fp32 reference numerics.
// ---------------------------------------------------------------------------

typedef __attribute__((ext_vector_type(16))) _Float16 v16h;
typedef __attribute__((ext_vector_type(8)))  float    v8f;

// ---------------------------- elementwise kernels --------------------------

__global__ void zero_kernel(float* __restrict__ p, long long n) {
  long long i  = (long long)blockIdx.x * blockDim.x + threadIdx.x;
  long long st = (long long)gridDim.x * blockDim.x;
  for (; i < n; i += st) p[i] = 0.0f;
}

__global__ void relu_kernel(float* __restrict__ p, long long n) {
  long long i  = (long long)blockIdx.x * blockDim.x + threadIdx.x;
  long long st = (long long)gridDim.x * blockDim.x;
  for (; i < n; i += st) p[i] = fmaxf(p[i], 0.0f);
}

// dst = relu(src * inv)   (HeteroConv mean over edge types + ReLU)
__global__ void combine_relu_kernel(float* __restrict__ dst, const float* __restrict__ src,
                                    long long n, float inv) {
  long long i  = (long long)blockIdx.x * blockDim.x + threadIdx.x;
  long long st = (long long)gridDim.x * blockDim.x;
  for (; i < n; i += st) dst[i] = fmaxf(src[i] * inv, 0.0f);
}

// ------------------------------ scatter (segment sum) ----------------------

__global__ void scatter_kernel(const float* __restrict__ x, const int* __restrict__ src,
                               const int* __restrict__ dst, float* __restrict__ agg,
                               float* __restrict__ cnt, int E, int F) {
  long long total = (long long)E * F;
  long long i  = (long long)blockIdx.x * blockDim.x + threadIdx.x;
  long long st = (long long)gridDim.x * blockDim.x;
  for (; i < total; i += st) {
    int e = (int)(i / F);
    int f = (int)(i - (long long)e * F);
    int s = src[e];
    int d = dst[e];
    atomicAdd(agg + (long long)d * F + f, x[(long long)s * F + f]);
    if (f == 0) atomicAdd(cnt + d, 1.0f);
  }
}

// agg[r, :] /= max(cnt[r], 1)
__global__ void mean_kernel(float* __restrict__ agg, const float* __restrict__ cnt,
                            int N, int F) {
  long long total = (long long)N * F;
  long long i  = (long long)blockIdx.x * blockDim.x + threadIdx.x;
  long long st = (long long)gridDim.x * blockDim.x;
  for (; i < total; i += st) {
    int r = (int)(i / F);
    float c = cnt[r];
    agg[i] *= 1.0f / fmaxf(c, 1.0f);
  }
}

// ------------------------------ WMMA GEMM ----------------------------------
// C[M,256] += A[M,K] @ W[K,256] (+ bias[256] once), fp32 in/out.
// A,W converted on the fly to split f16 (hi + lo); each 16x16x32 step issues
// 3x v_wmma_f32_16x16x32_f16 (hi*hi + lo*hi + hi*lo) -> ~fp32 accuracy.
// Block = 128 threads = 4 waves; each wave owns one 16x16 output tile.
// Requires M % 16 == 0 and K % 32 == 0 (true for all shapes here), so EXEC
// is all-ones for every WMMA (ISA requirement).

__global__ void gemm_wmma_kernel(const float* __restrict__ A, const float* __restrict__ W,
                                 const float* __restrict__ bias, float* __restrict__ C,
                                 int M, int K) {
  const int lane = threadIdx.x & 31;
  const int wid  = threadIdx.x >> 5;
  const int tile = blockIdx.x * 4 + wid;        // 16-row tile index
  if (tile * 16 >= M) return;                   // wave-uniform exit
  const int m0   = tile * 16;
  const int n0   = blockIdx.y * 16;             // 16-col tile (gridDim.y = 16)
  const int l16  = lane & 15;
  const int half = lane >> 4;                   // 0: lanes 0-15, 1: lanes 16-31

  // ---- load current C (+ bias once): lane holds col n0+l16, rows r+8*half
  v8f acc;
#pragma unroll
  for (int r = 0; r < 8; ++r) {
    float c0 = C[(long long)(m0 + r + half * 8) * 256 + n0 + l16];
    if (bias) c0 += bias[n0 + l16];
    acc[r] = c0;
  }

  const float* Arow = A + (long long)(m0 + l16) * K;   // A row M = m0 + lane%16

  for (int k0 = 0; k0 < K; k0 += 32) {
    v16h ahi, alo, bhi, blo;
    // A 16x32 f16 layout: lane row M = m0+l16; elems i -> K = k0+16*half+i
    // (same contiguous K order as B)
    const int ab = k0 + half * 16;
#pragma unroll
    for (int i = 0; i < 8; ++i) {
      float x = Arow[ab + i];
      _Float16 h = (_Float16)x;
      ahi[i] = h;
      alo[i] = (_Float16)(x - (float)h);
    }
#pragma unroll
    for (int i = 0; i < 8; ++i) {
      float x = Arow[ab + 8 + i];
      _Float16 h = (_Float16)x;
      ahi[8 + i] = h;
      alo[8 + i] = (_Float16)(x - (float)h);
    }
    // B 32x16 f16 layout: lane col N = n0+l16; elems i -> K = k0+16*half+i
    const int bb = k0 + half * 16;
#pragma unroll
    for (int i = 0; i < 16; ++i) {
      float x = W[(long long)(bb + i) * 256 + n0 + l16];
      _Float16 h = (_Float16)x;
      bhi[i] = h;
      blo[i] = (_Float16)(x - (float)h);
    }
    // D = A*B + C with split-precision correction terms
    acc = __builtin_amdgcn_wmma_f32_16x16x32_f16(false, ahi, false, bhi, (short)0, acc, false, false);
    acc = __builtin_amdgcn_wmma_f32_16x16x32_f16(false, alo, false, bhi, (short)0, acc, false, false);
    acc = __builtin_amdgcn_wmma_f32_16x16x32_f16(false, ahi, false, blo, (short)0, acc, false, false);
  }

#pragma unroll
  for (int r = 0; r < 8; ++r)
    C[(long long)(m0 + r + half * 8) * 256 + n0 + l16] = acc[r];
}

// ------------------------------- host side ---------------------------------

static inline int nblk(long long n, int bs, int cap) {
  long long b = (n + bs - 1) / bs;
  return (int)(b < cap ? b : cap);
}

extern "C" void kernel_launch(void* const* d_in, const int* in_sizes, int n_in,
                              void* d_out, int out_size, void* d_ws, size_t ws_size,
                              hipStream_t stream) {
  (void)out_size; (void)ws_size;
  if (n_in < 73) return;

  const int Nn[3] = {50000, 100000, 2000};     // user, ad, location
  const int Fr[3] = {64, 128, 32};
  const float* xr[3] = {(const float*)d_in[0], (const float*)d_in[1], (const float*)d_in[2]};

  struct Edge { int si, di, st, dt, E; };
  const Edge edges[6] = {
    {3, 4, 0, 1, 300000},   // click:        user -> ad
    {5, 6, 0, 1, 150000},   // purchase:     user -> ad
    {7, 8, 1, 0, 300000},   // rev_click:    ad -> user
    {9, 10, 1, 0, 150000},  // rev_purchase: ad -> user
    {11, 12, 1, 2, 100000}, // in:           ad -> location
    {13, 14, 2, 1, 100000}, // contains:     location -> ad
  };
  const float edgeTypeInv[3] = {1.0f / 2.0f, 1.0f / 3.0f, 1.0f};  // per dst type

  // --- params: detect flattening order from sizes.
  // insertion order: conv1.click.{Wl,bl,Wr}...  -> in_sizes[16] == 256 (bl)
  // jax-sorted:      conv1.click.{Wl,Wr,bl}...  -> in_sizes[16] == 32768 (Wr)
  const float *Wl[3][6], *bl[3][6], *Wr[3][6];
  const float *l1w, *l1b, *l2w, *l2b;
  const bool insertion = (in_sizes[16] == 256);
  if (insertion) {
    for (int c = 0; c < 3; ++c)
      for (int e = 0; e < 6; ++e) {
        int base = 15 + c * 18 + e * 3;
        Wl[c][e] = (const float*)d_in[base + 0];
        bl[c][e] = (const float*)d_in[base + 1];
        Wr[c][e] = (const float*)d_in[base + 2];
      }
    l1w = (const float*)d_in[69]; l1b = (const float*)d_in[70];
    l2w = (const float*)d_in[71]; l2b = (const float*)d_in[72];
  } else {
    // alphabetical edge order: click, contains, in, purchase, rev_click, rev_purchase
    const int alpha[6] = {0, 3, 4, 5, 2, 1};   // EDGE_LIST idx -> alphabetical pos
    for (int c = 0; c < 3; ++c)
      for (int e = 0; e < 6; ++e) {
        int base = 15 + c * 18 + alpha[e] * 3;
        Wl[c][e] = (const float*)d_in[base + 0];
        Wr[c][e] = (const float*)d_in[base + 1];
        bl[c][e] = (const float*)d_in[base + 2];
      }
    l1b = (const float*)d_in[69]; l1w = (const float*)d_in[70];
    l2b = (const float*)d_in[71]; l2w = (const float*)d_in[72];
  }

  // --- workspace partition (floats)
  float* ws = (float*)d_ws;
  size_t off = 0;
  float* cur[3]; float* nxt[3];
  for (int t = 0; t < 3; ++t) { cur[t] = ws + off; off += (size_t)Nn[t] * 256; }
  for (int t = 0; t < 3; ++t) { nxt[t] = ws + off; off += (size_t)Nn[t] * 256; }
  float* agg = ws + off; off += (size_t)100000 * 256;   // max N_dst * max F_src
  float* cnt = ws + off; off += 100000;

  auto zero = [&](float* p, long long n) {
    zero_kernel<<<nblk(n, 256, 16384), 256, 0, stream>>>(p, n);
  };
  auto gemm = [&](const float* A, const float* W, const float* bias, float* C,
                  int M, int K) {
    dim3 grid((M + 63) / 64, 16, 1);
    gemm_wmma_kernel<<<grid, dim3(128, 1, 1), 0, stream>>>(A, W, bias, C, M, K);
  };

  // --- 3 hetero SAGE layers
  for (int layer = 0; layer < 3; ++layer) {
    const bool first = (layer == 0);
    for (int t = 0; t < 3; ++t) zero(nxt[t], (long long)Nn[t] * 256);

    for (int e = 0; e < 6; ++e) {
      const Edge& ed = edges[e];
      const float* xs = first ? xr[ed.st] : cur[ed.st];
      const float* xd = first ? xr[ed.dt] : cur[ed.dt];
      const int Fs = first ? Fr[ed.st] : 256;
      const int Fd = first ? Fr[ed.dt] : 256;
      const int Nd = Nn[ed.dt];

      zero(agg, (long long)Nd * Fs);
      zero(cnt, Nd);
      long long tot = (long long)ed.E * Fs;
      scatter_kernel<<<nblk(tot, 256, 65536), 256, 0, stream>>>(
          xs, (const int*)d_in[ed.si], (const int*)d_in[ed.di], agg, cnt, ed.E, Fs);
      mean_kernel<<<nblk((long long)Nd * Fs, 256, 16384), 256, 0, stream>>>(agg, cnt, Nd, Fs);

      gemm(agg, Wl[layer][e], bl[layer][e], nxt[ed.dt], Nd, Fs);   // += mean @ Wl + bl
      gemm(xd, Wr[layer][e], nullptr, nxt[ed.dt], Nd, Fd);         // += x_dst @ Wr
    }
    for (int t = 0; t < 3; ++t)
      combine_relu_kernel<<<nblk((long long)Nn[t] * 256, 256, 16384), 256, 0, stream>>>(
          cur[t], nxt[t], (long long)Nn[t] * 256, edgeTypeInv[t]);
  }

  // --- MLP head: out = (relu(x @ lin1 + b1)) @ lin2 + b2, concat u/ad/loc
  float* outp = (float*)d_out;
  zero(outp, (long long)(50000 + 100000 + 2000) * 256);
  float* tmp = agg;  // 100000*256 floats, large enough for any node type
  long long rowOff = 0;
  for (int t = 0; t < 3; ++t) {
    zero(tmp, (long long)Nn[t] * 256);
    gemm(cur[t], l1w, l1b, tmp, Nn[t], 256);
    relu_kernel<<<nblk((long long)Nn[t] * 256, 256, 16384), 256, 0, stream>>>(
        tmp, (long long)Nn[t] * 256);
    gemm(tmp, l2w, l2b, outp + rowOff * 256, Nn[t], 256);
    rowOff += Nn[t];
  }
}
